// ReconfiguredAttentionModule_75488345194994
// MI455X (gfx1250) — matrix operand order repaired and checked by reading, not compiled, not run on hardware
//
#include <hip/hip_runtime.h>
#include <stdint.h>

typedef __attribute__((ext_vector_type(16))) __bf16 v16bf;
typedef __attribute__((ext_vector_type(8)))  float  v8f;

#define NH    8
#define HD    32
#define CDIM  256
#define BATCH 32
#define HDIM  56
#define NTOK  3136
#define NDET  100
#define TTOK  3236
#define MROWS (BATCH*TTOK)   /* 103552 */
#define QKVN  768
#define LWIN  49
#define ATT_SCALE 0.17677669529663689f

union FragU { uint4 q[2]; v16bf v; };

__device__ __forceinline__ uint16_t f2bf(float f) {
  union { float f; uint32_t u; } v; v.f = f;
  uint32_t u = v.u, r;
  if ((u & 0x7F800000u) == 0x7F800000u) r = u;            // Inf/NaN passthrough
  else r = u + 0x7FFFu + ((u >> 16) & 1u);                // round-nearest-even
  return (uint16_t)(r >> 16);
}

__device__ __forceinline__ v8f zero8() {
  v8f z;
#pragma unroll
  for (int i = 0; i < 8; ++i) z[i] = 0.0f;
  return z;
}

__device__ __forceinline__ v16bf zero_frag() {
  FragU z; z.q[0] = make_uint4(0,0,0,0); z.q[1] = make_uint4(0,0,0,0); return z.v;
}

// A fragment (16x32 bf16), lane&15 = row.
// lanes 0-15 hold K {0..7, 16..23}; lanes 16-31 hold K {8..15, 24..31}.
__device__ __forceinline__ v16bf load_a(const uint16_t* rowp, int half) {
  FragU u;
  const uint16_t* p = rowp + (half ? 8 : 0);
  u.q[0] = *(const uint4*)(p);
  u.q[1] = *(const uint4*)(p + 16);
  return u.v;
}

// B fragment (32x16 bf16), lane&15 = column (same K order as A: lanes 0-15
// K {0..7, 16..23}; lanes 16-31 K {8..15, 24..31}). colp points at the
// column's K-major data (contiguous along K).
__device__ __forceinline__ v16bf load_b(const uint16_t* colp, int half) {
  FragU u;
  const uint16_t* p = colp + (half ? 8 : 0);
  u.q[0] = *(const uint4*)(p);
  u.q[1] = *(const uint4*)(p + 16);
  return u.v;
}

__device__ __forceinline__ v8f wmma_bf16(v16bf a, v16bf b, v8f c) {
  return __builtin_amdgcn_wmma_f32_16x16x32_bf16(false, a, false, b, (short)0, c, false, false);
}

// ---------------------------------------------------------------- cast kernels
__global__ void cast_tokens_k(const float* __restrict__ x, const float* __restrict__ det,
                              uint16_t* __restrict__ tok) {
  size_t i = (size_t)blockIdx.x * blockDim.x + threadIdx.x;
  size_t stride = (size_t)gridDim.x * blockDim.x;
  const size_t nx = (size_t)BATCH * NTOK * CDIM;
  for (size_t p = i; p < nx; p += stride) {
    size_t b = p / ((size_t)NTOK * CDIM);
    size_t r = p - b * ((size_t)NTOK * CDIM);
    tok[b * ((size_t)TTOK * CDIM) + r] = f2bf(x[p]);
  }
  const size_t nd = (size_t)BATCH * NDET * CDIM;
  for (size_t p = i; p < nd; p += stride) {
    size_t b = p / ((size_t)NDET * CDIM);
    size_t r = p - b * ((size_t)NDET * CDIM);
    tok[b * ((size_t)TTOK * CDIM) + (size_t)NTOK * CDIM + r] = f2bf(det[p]);
  }
}

__global__ void cast_weights_k(const float* __restrict__ wq, const float* __restrict__ wp,
                               uint16_t* __restrict__ oq, uint16_t* __restrict__ op) {
  int i = blockIdx.x * blockDim.x + threadIdx.x;
  int st = gridDim.x * blockDim.x;
  for (int p = i; p < QKVN * CDIM; p += st) oq[p] = f2bf(wq[p]);
  for (int p = i; p < CDIM * CDIM; p += st) op[p] = f2bf(wp[p]);
}

// ------------------------------------------------------- GEMM: out = A @ W^T + b
// A: MROWS x 256 bf16 row-major.  W: Ncols x 256 bf16 row-major (so B[k][n] =
// W[n][k]; B column n == W row n, contiguous along K -> direct load_b).
// Wave computes a 16(M) x 64(N) strip. grid = (Ncols/64, MROWS/128), block 256.
__global__ __launch_bounds__(256) void gemm_bias_bf16_k(
    const uint16_t* __restrict__ A, const uint16_t* __restrict__ W,
    const float* __restrict__ bias, uint16_t* __restrict__ Out, int Ncols) {
  const int lane = threadIdx.x & 31, half = lane >> 4, ln = lane & 15;
  const int wave = threadIdx.x >> 5;
  const int m0 = (blockIdx.y * 8 + wave) * 16;
  const int n0 = blockIdx.x * 64;

  v8f acc[4] = { zero8(), zero8(), zero8(), zero8() };
  const uint16_t* arow = A + (size_t)(m0 + ln) * CDIM;

  for (int k0 = 0; k0 < CDIM; k0 += 32) {
    if (k0 + 32 < CDIM) __builtin_prefetch(arow + k0 + 32, 0, 1);
    v16bf a = load_a(arow + k0, half);
#pragma unroll
    for (int j = 0; j < 4; ++j) {
      const uint16_t* colp = W + (size_t)(n0 + j * 16 + ln) * CDIM + k0;
      acc[j] = wmma_bf16(a, load_b(colp, half), acc[j]);
    }
  }
#pragma unroll
  for (int j = 0; j < 4; ++j) {
    int col = n0 + j * 16 + ln;
    float bv = bias[col];
#pragma unroll
    for (int r = 0; r < 8; ++r) {
      int row = m0 + r + half * 8;
      Out[(size_t)row * Ncols + col] = f2bf(acc[j][r] + bv);
    }
  }
}

// ------------------------------------------------------- window attention
// One block per window (2048), one wave per head. qkv row = b*TTOK + token,
// layout per token: [q(8x32) | k(8x32) | v(8x32)].
__global__ __launch_bounds__(256) void win_attn_k(
    const uint16_t* __restrict__ qkv, const float* __restrict__ maskG,
    const float* __restrict__ biasTab, uint16_t* __restrict__ outBf) {
  __shared__ __align__(16) uint16_t sK [NH][64 * 32];  // keys x d
  __shared__ __align__(16) uint16_t sVt[NH][32 * 64];  // d x keys
  __shared__ __align__(16) uint16_t sP [NH][16 * 64];  // probs bounce
  __shared__ float sBias[169 * NH];
  __shared__ float sMask[LWIN * LWIN];

  const int wi = blockIdx.x;
  const int b = wi >> 6, win = wi & 63;
  const int wr = win >> 3, wc = win & 7;
  const int tid = threadIdx.x;
  const int lane = tid & 31, half = lane >> 4, ln = lane & 15;
  const int h = tid >> 5;

  for (int i = tid; i < 169 * NH; i += 256) sBias[i] = biasTab[i];
  for (int i = tid; i < LWIN * LWIN; i += 256) sMask[i] = maskG[(size_t)win * LWIN * LWIN + i];

  for (int l = lane; l < 64; l += 32) {
    if (l < LWIN) {
      int lr = l / 7, lc = l - lr * 7;
      int t = (wr * 7 + lr) * HDIM + (wc * 7 + lc);
      const uint16_t* kp = qkv + ((size_t)(b * TTOK + t)) * QKVN + CDIM + h * HD;
      const uint16_t* vp = kp + CDIM;
      *(uint4*)&sK[h][l * 32 +  0] = *(const uint4*)(kp +  0);
      *(uint4*)&sK[h][l * 32 +  8] = *(const uint4*)(kp +  8);
      *(uint4*)&sK[h][l * 32 + 16] = *(const uint4*)(kp + 16);
      *(uint4*)&sK[h][l * 32 + 24] = *(const uint4*)(kp + 24);
#pragma unroll
      for (int d = 0; d < HD; ++d) sVt[h][d * 64 + l] = vp[d];
    } else {
      uint4 z = make_uint4(0,0,0,0);
      *(uint4*)&sK[h][l * 32 +  0] = z;
      *(uint4*)&sK[h][l * 32 +  8] = z;
      *(uint4*)&sK[h][l * 32 + 16] = z;
      *(uint4*)&sK[h][l * 32 + 24] = z;
#pragma unroll
      for (int d = 0; d < HD; ++d) sVt[h][d * 64 + l] = 0;
    }
  }
  __syncthreads();

  for (int mt = 0; mt < 4; ++mt) {
    const int q0 = mt * 16;
    v16bf aq;
    {
      int lq = q0 + ln;
      if (lq < LWIN) {
        int lr = lq / 7, lc = lq - lr * 7;
        int t = (wr * 7 + lr) * HDIM + (wc * 7 + lc);
        aq = load_a(qkv + ((size_t)(b * TTOK + t)) * QKVN + h * HD, half);
      } else aq = zero_frag();
    }
    v8f S[4];
#pragma unroll
    for (int j = 0; j < 4; ++j)
      S[j] = wmma_bf16(aq, load_b(&sK[h][(j * 16 + ln) * 32], half), zero8());

    // row-wise softmax with bias + mask; padded keys -> 0 probability
#pragma unroll
    for (int r = 0; r < 8; ++r) {
      int q = q0 + r + half * 8;
      float lv[4];
#pragma unroll
      for (int j = 0; j < 4; ++j) {
        int k = j * 16 + ln;
        float t = -3.0e38f;
        if (q < LWIN && k < LWIN) {
          int qr = q / 7, qc = q - qr * 7;
          int kr = k / 7, kc = k - kr * 7;
          int ridx = (qr - kr + 6) * 13 + (qc - kc + 6);
          t = S[j][r] * ATT_SCALE + sBias[ridx * NH + h] + sMask[q * LWIN + k];
        }
        lv[j] = t;
      }
      float m = fmaxf(fmaxf(lv[0], lv[1]), fmaxf(lv[2], lv[3]));
#pragma unroll
      for (int d2 = 1; d2 < 16; d2 <<= 1) m = fmaxf(m, __shfl_xor(m, d2, 32));
      float e[4]; float s = 0.0f;
#pragma unroll
      for (int j = 0; j < 4; ++j) { e[j] = (lv[j] > -1.0e38f) ? __expf(lv[j] - m) : 0.0f; s += e[j]; }
#pragma unroll
      for (int d2 = 1; d2 < 16; d2 <<= 1) s += __shfl_xor(s, d2, 32);
      float inv = (s > 0.0f) ? (1.0f / s) : 0.0f;
#pragma unroll
      for (int j = 0; j < 4; ++j)
        sP[h][(r + half * 8) * 64 + (j * 16 + ln)] = f2bf(e[j] * inv);
    }
    __syncthreads();

    v8f O[2] = { zero8(), zero8() };
#pragma unroll
    for (int ks = 0; ks < 2; ++ks) {
      v16bf ap = load_a(&sP[h][ln * 64 + ks * 32], half);
#pragma unroll
      for (int jn = 0; jn < 2; ++jn)
        O[jn] = wmma_bf16(ap, load_b(&sVt[h][(jn * 16 + ln) * 64 + ks * 32], half), O[jn]);
    }
#pragma unroll
    for (int jn = 0; jn < 2; ++jn) {
      int col = h * HD + jn * 16 + ln;
#pragma unroll
      for (int r = 0; r < 8; ++r) {
        int lo = q0 + r + half * 8;
        if (lo < LWIN) {
          int lr = lo / 7, lc = lo - lr * 7;
          int t = (wr * 7 + lr) * HDIM + (wc * 7 + lc);
          outBf[((size_t)(b * TTOK + t)) * CDIM + col] = f2bf(O[jn][r]);
        }
      }
    }
    __syncthreads();
  }
}

// ------------------------------------------------------- det-token attention
// grid (2, 32): 4 heads per block (blockDim 128), batch = blockIdx.y.
__global__ __launch_bounds__(128) void det_attn_k(const uint16_t* __restrict__ qkv,
                                                  uint16_t* __restrict__ outBf) {
  __shared__ __align__(16) uint16_t dK [4][128 * 32];
  __shared__ __align__(16) uint16_t dVt[4][32 * 128];
  __shared__ __align__(16) uint16_t dP [4][16 * 128];
  const int b = blockIdx.y;
  const int wave = threadIdx.x >> 5;
  const int h = blockIdx.x * 4 + wave;
  const int lane = threadIdx.x & 31, half = lane >> 4, ln = lane & 15;

  for (int l = lane; l < 128; l += 32) {
    if (l < NDET) {
      const uint16_t* kp = qkv + ((size_t)(b * TTOK + NTOK + l)) * QKVN + CDIM + h * HD;
      const uint16_t* vp = kp + CDIM;
      *(uint4*)&dK[wave][l * 32 +  0] = *(const uint4*)(kp +  0);
      *(uint4*)&dK[wave][l * 32 +  8] = *(const uint4*)(kp +  8);
      *(uint4*)&dK[wave][l * 32 + 16] = *(const uint4*)(kp + 16);
      *(uint4*)&dK[wave][l * 32 + 24] = *(const uint4*)(kp + 24);
#pragma unroll
      for (int d = 0; d < HD; ++d) dVt[wave][d * 128 + l] = vp[d];
    } else {
      uint4 z = make_uint4(0,0,0,0);
      *(uint4*)&dK[wave][l * 32 +  0] = z;
      *(uint4*)&dK[wave][l * 32 +  8] = z;
      *(uint4*)&dK[wave][l * 32 + 16] = z;
      *(uint4*)&dK[wave][l * 32 + 24] = z;
#pragma unroll
      for (int d = 0; d < HD; ++d) dVt[wave][d * 128 + l] = 0;
    }
  }
  __syncthreads();

  for (int mt = 0; mt < 7; ++mt) {
    const int q0 = mt * 16;
    v16bf aq;
    {
      int lq = q0 + ln;
      if (lq < NDET)
        aq = load_a(qkv + ((size_t)(b * TTOK + NTOK + lq)) * QKVN + h * HD, half);
      else aq = zero_frag();
    }
    v8f S[8];
#pragma unroll
    for (int j = 0; j < 8; ++j)
      S[j] = wmma_bf16(aq, load_b(&dK[wave][(j * 16 + ln) * 32], half), zero8());

#pragma unroll
    for (int r = 0; r < 8; ++r) {
      int q = q0 + r + half * 8;
      float lv[8];
#pragma unroll
      for (int j = 0; j < 8; ++j) {
        int k = j * 16 + ln;
        lv[j] = (q < NDET && k < NDET) ? S[j][r] * ATT_SCALE : -3.0e38f;
      }
      float m = lv[0];
#pragma unroll
      for (int j = 1; j < 8; ++j) m = fmaxf(m, lv[j]);
#pragma unroll
      for (int d2 = 1; d2 < 16; d2 <<= 1) m = fmaxf(m, __shfl_xor(m, d2, 32));
      float e[8]; float s = 0.0f;
#pragma unroll
      for (int j = 0; j < 8; ++j) { e[j] = (lv[j] > -1.0e38f) ? __expf(lv[j] - m) : 0.0f; s += e[j]; }
#pragma unroll
      for (int d2 = 1; d2 < 16; d2 <<= 1) s += __shfl_xor(s, d2, 32);
      float inv = (s > 0.0f) ? (1.0f / s) : 0.0f;
#pragma unroll
      for (int j = 0; j < 8; ++j)
        dP[wave][(r + half * 8) * 128 + (j * 16 + ln)] = f2bf(e[j] * inv);
    }
    __syncthreads();

    v8f O[2] = { zero8(), zero8() };
#pragma unroll
    for (int ks = 0; ks < 4; ++ks) {
      v16bf ap = load_a(&dP[wave][ln * 128 + ks * 32], half);
#pragma unroll
      for (int jn = 0; jn < 2; ++jn)
        O[jn] = wmma_bf16(ap, load_b(&dVt[wave][(jn * 16 + ln) * 128 + ks * 32], half), O[jn]);
    }
#pragma unroll
    for (int jn = 0; jn < 2; ++jn) {
      int col = h * HD + jn * 16 + ln;
#pragma unroll
      for (int r = 0; r < 8; ++r) {
        int lo = q0 + r + half * 8;
        if (lo < NDET)
          outBf[((size_t)(b * TTOK + NTOK + lo)) * CDIM + col] = f2bf(O[jn][r]);
      }
    }
    __syncthreads();
  }
}

// ------------------------------------------------------- final projection
// out = aout @ proj_w^T + proj_b, scattered into (patch, det) concat layout.
__global__ __launch_bounds__(256) void gemm_proj_k(
    const uint16_t* __restrict__ A, const uint16_t* __restrict__ W,
    const float* __restrict__ bias, float* __restrict__ out) {
  const int lane = threadIdx.x & 31, half = lane >> 4, ln = lane & 15;
  const int wave = threadIdx.x >> 5;
  const int m0 = (blockIdx.y * 8 + wave) * 16;
  const int n0 = blockIdx.x * 64;

  v8f acc[4] = { zero8(), zero8(), zero8(), zero8() };
  const uint16_t* arow = A + (size_t)(m0 + ln) * CDIM;
  for (int k0 = 0; k0 < CDIM; k0 += 32) {
    if (k0 + 32 < CDIM) __builtin_prefetch(arow + k0 + 32, 0, 1);
    v16bf a = load_a(arow + k0, half);
#pragma unroll
    for (int j = 0; j < 4; ++j)
      acc[j] = wmma_bf16(a, load_b(W + (size_t)(n0 + j * 16 + ln) * CDIM + k0, half), acc[j]);
  }
  const size_t detBase = (size_t)BATCH * NTOK * CDIM;
#pragma unroll
  for (int j = 0; j < 4; ++j) {
    int col = n0 + j * 16 + ln;
    float bv = bias[col];
#pragma unroll
    for (int r = 0; r < 8; ++r) {
      int g = m0 + r + half * 8;
      int b = g / TTOK;
      int t = g - b * TTOK;
      size_t idx = (t < NTOK)
          ? ((size_t)(b * NTOK + t) * CDIM + col)
          : (detBase + (size_t)(b * NDET + (t - NTOK)) * CDIM + col);
      out[idx] = acc[j][r] + bv;
    }
  }
}

// ---------------------------------------------------------------- launch
extern "C" void kernel_launch(void* const* d_in, const int* in_sizes, int n_in,
                              void* d_out, int out_size, void* d_ws, size_t ws_size,
                              hipStream_t stream) {
  const float* x     = (const float*)d_in[0];
  const float* det   = (const float*)d_in[1];
  const float* mask  = (const float*)d_in[2];
  const float* qkvw  = (const float*)d_in[3];
  const float* qkvb  = (const float*)d_in[4];
  const float* projw = (const float*)d_in[5];
  const float* projb = (const float*)d_in[6];
  const float* btab  = (const float*)d_in[7];
  float* out = (float*)d_out;

  const size_t TOK_B = (size_t)MROWS * CDIM * 2;   // 53,018,624
  const size_t WQ_B  = (size_t)QKVN * CDIM * 2;    //    393,216
  const size_t WP_B  = (size_t)CDIM * CDIM * 2;    //    131,072
  const size_t QKV_B = (size_t)MROWS * QKVN * 2;   // 159,055,872
  const size_t AO_B  = (size_t)MROWS * CDIM * 2;   // 53,018,624
  if (ws_size < TOK_B + WQ_B + WP_B + QKV_B + AO_B) return;

  char* ws = (char*)d_ws;
  uint16_t* tok  = (uint16_t*)(ws);
  uint16_t* wq   = (uint16_t*)(ws + TOK_B);
  uint16_t* wp   = (uint16_t*)(ws + TOK_B + WQ_B);
  uint16_t* qkvB = (uint16_t*)(ws + TOK_B + WQ_B + WP_B);
  uint16_t* aout = (uint16_t*)(ws + TOK_B + WQ_B + WP_B + QKV_B);

  cast_tokens_k <<<dim3(4096), dim3(256), 0, stream>>>(x, det, tok);
  cast_weights_k<<<dim3(256),  dim3(256), 0, stream>>>(qkvw, projw, wq, wp);
  gemm_bias_bf16_k<<<dim3(12, 809), dim3(256), 0, stream>>>(tok, wq, qkvb, qkvB, QKVN);
  win_attn_k<<<dim3(BATCH * 64), dim3(256), 0, stream>>>(qkvB, mask, btab, aout);
  det_attn_k<<<dim3(2, BATCH), dim3(128), 0, stream>>>(qkvB, aout);
  gemm_proj_k<<<dim3(4, 809), dim3(256), 0, stream>>>(aout, wp, projb, out);
}
